// MemoryReader_81484119540553
// MI455X (gfx1250) — matrix operand order repaired and checked by reading, not compiled, not run on hardware
//
#include <hip/hip_runtime.h>
#include <hip/hip_bf16.h>
#include <math.h>

// ---------------------------------------------------------------------------
// MemoryReader fused kernel for gfx1250 (MI455X, wave32, WMMA).
//   mk: [4, 64, 8192]  (keys of memory tokens, flattened)
//   qk: [4, 64, 1024]  (query keys)
//   mv: [4, 512, 8192] (memory values)
//   out:[4, 512, 1024]
// Math: S = (2*mk^T qk - |mk|^2)/8 ; P = softmax_n(S) ; out = mv @ P
// All GEMMs via V_WMMA_F32_16X16X32_BF16 using a bf16 hi/lo split
// (hi*hi + hi*lo + lo*hi, f32 accumulate) for near-fp32 accuracy.
// Scale 1/sqrt(64) is folded into the qk (x0.25) and |mk|^2 (x0.125) loads.
// ---------------------------------------------------------------------------

typedef __attribute__((ext_vector_type(16))) __bf16 v16bf;
typedef __attribute__((ext_vector_type(8)))  float  v8f;

union BF16Frag {
    v16bf v;
    unsigned short u[16];
    uint4 q[2];
};

static __device__ __forceinline__ v8f wmma_bf16(v16bf a, v16bf b, v8f c) {
    // (neg_a, A, neg_b, B, c_mod, C, reuse_a, reuse_b)
    return __builtin_amdgcn_wmma_f32_16x16x32_bf16(false, a, false, b,
                                                   (short)0, c, false, false);
}

__global__ __launch_bounds__(256)
void memory_reader_fused(const float* __restrict__ mk,
                         const float* __restrict__ qk,
                         const float* __restrict__ mv,
                         float* __restrict__ out)
{
    constexpr int CK = 64, N = 8192, M = 1024, CV = 512;
    constexpr int NT    = 32;       // n-tile == WMMA K
    constexpr int ITERS = N / NT;   // 256
    constexpr int MKS   = 36;       // lds_mk row stride (floats), 16B aligned, bank-staggered
    constexpr int PS    = 40;       // P row stride (ushorts) = 80B, 16B aligned

    __shared__ float          lds_mk[CK * MKS];   // [ck][n] f32 tile (64 x 32)
    __shared__ unsigned short lds_phi[64 * PS];   // P hi bf16, [m][n]
    __shared__ unsigned short lds_plo[64 * PS];   // P lo bf16, [m][n]
    __shared__ float          lds_asq[NT];        // 0.125*|mk_n|^2 for this tile
    __shared__ float          lds_factor[64];     // per-column rescale exp(m_old-m_new)
    __shared__ float          lds_rowsum[64];     // final softmax denominators
    __shared__ int            lds_flag;           // 1 if any column max changed

    const int blk   = blockIdx.x;          // 256 blocks: b(4) x mt(16) x cvt(4)
    const int cvt   = blk & 3;
    const int mt    = (blk >> 2) & 15;
    const int b     = blk >> 6;
    const int tid   = threadIdx.x;
    const int wave  = tid >> 5;            // 8 waves
    const int lane  = tid & 31;
    const int l16   = lane & 15;
    const int lhalf = lane >> 4;

    const float* mkB = mk + (size_t)b * CK * N;
    const float* qkB = qk + (size_t)b * CK * M;
    const float* mvB = mv + (size_t)b * CV * N;

    const int  m0     = mt * 64;                 // 64 query columns per block
    const int  cv0    = cvt * 128 + wave * 16;   // 16 value channels per wave
    const bool stageA = (wave < 4);              // waves 0..3 own 16 columns each

    // ---- preload qk B-fragments (x0.25 folded in), hi/lo split, K=64 -> 2 ksteps
    BF16Frag qhi[2], qlo[2];
    if (stageA) {
        const int mcol = m0 + wave * 16 + l16;
        #pragma unroll
        for (int ks = 0; ks < 2; ++ks) {
            #pragma unroll
            for (int e = 0; e < 16; ++e) {
                const int ck = ks * 32 + lhalf * 16 + e;   // contiguous k order (same as A): lanes<16 K=0..15
                const float x = 0.25f * qkB[(size_t)ck * M + mcol];
                const __bf16 h = (__bf16)x;                // hw RN conversion
                qhi[ks].v[e] = h;
                qlo[ks].v[e] = (__bf16)(x - (float)h);
            }
        }
    }

    v8f acc[4];                                  // 16cv x 64m accumulators / wave
    #pragma unroll
    for (int j = 0; j < 4; ++j) acc[j] = (v8f)0.0f;

    float run_max = -INFINITY;
    float run_sum = 0.0f;

    const int ld_ck = tid >> 2;                  // cooperative mk loader mapping
    const int ld_nn = (tid & 3) * 8;

    for (int it = 0; it < ITERS; ++it) {
        const int n0 = it * NT;

        // Phase 1a: mv A-fragment loads straight into registers (A 16x32 layout)
        const float* pmv = mvB + (size_t)(cv0 + l16) * N + n0 + lhalf * 8;
        union { float4 v4[4]; float f[16]; } fa;
        fa.v4[0] = *(const float4*)(pmv);        // K = base .. base+3
        fa.v4[1] = *(const float4*)(pmv + 4);    // K = base+4 .. base+7
        fa.v4[2] = *(const float4*)(pmv + 16);   // K = base+16 .. base+19
        fa.v4[3] = *(const float4*)(pmv + 20);   // K = base+20 .. base+23

        // Phase 1b: cooperative mk tile (64ck x 32n) -> LDS
        const float* pmk = mkB + (size_t)ld_ck * N + n0 + ld_nn;
        *(float4*)&lds_mk[ld_ck * MKS + ld_nn]     = *(const float4*)(pmk);
        *(float4*)&lds_mk[ld_ck * MKS + ld_nn + 4] = *(const float4*)(pmk + 4);

        __syncthreads();

        // Phase 2: 0.125*|mk_n|^2 for the 32 tokens of this tile (wave 0)
        if (wave == 0) {
            if (lane == 0) lds_flag = 0;
            float s = 0.0f;
            #pragma unroll 8
            for (int ck = 0; ck < CK; ++ck) {
                const float v = lds_mk[ck * MKS + lane];
                s += v * v;
            }
            lds_asq[lane] = s * 0.125f;
        }
        __syncthreads();

        // Phase 3: logits via WMMA, online softmax, P(hi/lo) -> LDS
        if (stageA) {
            float sl[16];
            #pragma unroll
            for (int i = 0; i < 2; ++i) {        // two 16-row n-subtiles
                v8f c = (v8f)0.0f;
                #pragma unroll
                for (int ks = 0; ks < 2; ++ks) {
                    BF16Frag ahi, alo;           // A = mk^T, transposed gather from LDS
                    #pragma unroll
                    for (int e = 0; e < 16; ++e) {
                        const int ck = ks * 32 + lhalf * 16 + e;
                        const float x = lds_mk[ck * MKS + i * 16 + l16];
                        const __bf16 h = (__bf16)x;
                        ahi.v[e] = h;
                        alo.v[e] = (__bf16)(x - (float)h);
                    }
                    c = wmma_bf16(ahi.v, qhi[ks].v, c);
                    c = wmma_bf16(ahi.v, qlo[ks].v, c);
                    c = wmma_bf16(alo.v, qhi[ks].v, c);
                }
                #pragma unroll
                for (int r = 0; r < 8; ++r) {    // C layout: row = r + 8*lhalf
                    const int nloc = i * 16 + r + 8 * lhalf;
                    sl[i * 8 + r] = c[r] - lds_asq[nloc];
                }
            }

            float mloc = sl[0];
            #pragma unroll
            for (int t = 1; t < 16; ++t) mloc = fmaxf(mloc, sl[t]);
            mloc = fmaxf(mloc, __shfl_xor(mloc, 16, 32));   // combine lane halves

            const bool changed = (mloc > run_max);          // any column max update?
            if (__any(changed) && lane == 0) lds_flag = 1;

            const float m_new  = fmaxf(run_max, mloc);
            const float factor = __expf(run_max - m_new);   // first iter: exp(-inf)=0
            run_max = m_new;

            float psum = 0.0f;
            #pragma unroll
            for (int i = 0; i < 2; ++i) {
                #pragma unroll
                for (int r = 0; r < 8; ++r) {
                    const float p = __expf(sl[i * 8 + r] - m_new);
                    psum += p;
                    const int nloc = i * 16 + r + 8 * lhalf;
                    const int a = (wave * 16 + l16) * PS + nloc;
                    const __bf16 h = (__bf16)p;
                    lds_phi[a] = __builtin_bit_cast(unsigned short, h);
                    const __bf16 l = (__bf16)(p - (float)h);
                    lds_plo[a] = __builtin_bit_cast(unsigned short, l);
                }
            }
            psum += __shfl_xor(psum, 16, 32);
            run_sum = run_sum * factor + psum;
            if (lane < 16) lds_factor[wave * 16 + lane] = factor;
        }
        __syncthreads();

        // Phase 4: readout GEMM, all 8 waves (16cv x 64m each)
        const int needRescale = lds_flag;        // block-uniform
        if (needRescale) {                       // rare after warm-up
            #pragma unroll
            for (int j = 0; j < 4; ++j) {
                const float fac = lds_factor[j * 16 + l16];
                #pragma unroll
                for (int r = 0; r < 8; ++r) acc[j][r] *= fac;
            }
        }

        BF16Frag mvhi, mvlo;
        #pragma unroll
        for (int e = 0; e < 16; ++e) {
            const __bf16 h = (__bf16)fa.f[e];
            mvhi.v[e] = h;
            mvlo.v[e] = (__bf16)(fa.f[e] - (float)h);
        }
        #pragma unroll
        for (int j = 0; j < 4; ++j) {
            BF16Frag phi, plo;                   // B = P, [m][n] rows, 2x b128
            const int base = (j * 16 + l16) * PS + lhalf * 8;
            phi.q[0] = *(const uint4*)&lds_phi[base];
            phi.q[1] = *(const uint4*)&lds_phi[base + 16];
            plo.q[0] = *(const uint4*)&lds_plo[base];
            plo.q[1] = *(const uint4*)&lds_plo[base + 16];
            v8f a = acc[j];
            a = wmma_bf16(mvhi.v, phi.v, a);
            a = wmma_bf16(mvhi.v, plo.v, a);
            a = wmma_bf16(mvlo.v, phi.v, a);
            acc[j] = a;
        }
        // No barrier here: next iteration's two barriers already separate the
        // lds_mk/asq/P/flag rewrites from all readers of this iteration.
    }

    if (stageA && lane < 16) lds_rowsum[wave * 16 + lane] = run_sum;
    __syncthreads();

    // Normalize by softmax denominator and store out[b][cv][m]
    #pragma unroll
    for (int j = 0; j < 4; ++j) {
        const float inv = 1.0f / lds_rowsum[j * 16 + l16];
        #pragma unroll
        for (int r = 0; r < 8; ++r) {
            const int cv = cv0 + r + 8 * lhalf;
            const int m  = m0 + j * 16 + l16;
            out[((size_t)b * CV + cv) * M + m] = acc[j][r] * inv;
        }
    }
}

extern "C" void kernel_launch(void* const* d_in, const int* in_sizes, int n_in,
                              void* d_out, int out_size, void* d_ws, size_t ws_size,
                              hipStream_t stream) {
    (void)in_sizes; (void)n_in; (void)d_ws; (void)ws_size; (void)out_size;
    const float* mk = (const float*)d_in[0];
    const float* qk = (const float*)d_in[1];
    const float* mv = (const float*)d_in[2];
    float* out = (float*)d_out;
    // 256 blocks = 4 batches x 16 m-tiles(64) x 4 cv-chunks(128); 256 thr = 8 waves
    hipLaunchKernelGGL(memory_reader_fused, dim3(256), dim3(256), 0, stream,
                       mk, qk, mv, out);
}
